// SparseMoeWrapper_1726576855474
// MI455X (gfx1250) — compile-verified, matrix operand order checked by reading, not run on hardware
//
#include <hip/hip_runtime.h>
#include <hip/hip_bf16.h>

typedef __attribute__((ext_vector_type(16))) __bf16 v16bf;
typedef __attribute__((ext_vector_type(8)))  __bf16 v8bf;
typedef __attribute__((ext_vector_type(8)))  float  v8f;

#define T_TOK 4096
#define DDIM  1024
#define FDIM  4096
#define NEXP  8
#define M_TILE 32

// workspace layout (bytes)
#define COMB_OFF  0u                          // float[NEXP*T_TOK]   = 131072 B
#define CNT_OFF   131072u                     // int[NEXP]
#define OFFS_OFF  (CNT_OFF + 64u)             // int[NEXP]
#define LIST_OFF  (CNT_OFF + 128u)            // int[NEXP*T_TOK]     = 131072 B
#define HBUF_OFF  (LIST_OFF + 131072u + 128u) // __bf16[2*T_TOK*FDIM] = 64 MB

__device__ __forceinline__ v16bf make_frag(const __bf16* lo, const __bf16* hi) {
    v8bf l = *(const v8bf*)lo;
    v8bf h = *(const v8bf*)hi;
    v16bf r;
#pragma unroll
    for (int i = 0; i < 8; ++i) { r[i] = l[i]; r[i + 8] = h[i]; }
    return r;
}

// silu(g)*u with fast v_rcp_f32 instead of IEEE divide chain
__device__ __forceinline__ float fast_silu_mul(float g, float u) {
    return g * __builtin_amdgcn_rcpf(1.0f + __expf(-g)) * u;
}

// stage a 32(K) x 256(N) f32 chunk into LDS transposed [256][32] bf16.
// Two K-rows per thread-step -> packed 2xbf16 dword store (v_cvt_pk_bf16_f32).
__device__ __forceinline__ void stage_b_t(const float* __restrict__ src, int ldsrc,
                                          __bf16 (*dst)[32], int tid) {
#pragma unroll
    for (int j = 0; j < 16; ++j) {
        int idx = tid + 256 * j;
        int n   = idx & 255;
        int k2  = idx >> 8;                 // 0..15 -> K rows 2*k2, 2*k2+1
        float a = src[(size_t)(2 * k2) * ldsrc + n];
        float b = src[(size_t)(2 * k2 + 1) * ldsrc + n];
        union { unsigned int u; __bf16 bf[2]; } pk;
        pk.bf[0] = (__bf16)a;
        pk.bf[1] = (__bf16)b;
        *reinterpret_cast<unsigned int*>(&dst[n][2 * k2]) = pk.u;
    }
}

// one global_prefetch_b8 per thread covers the whole next 32x256 f32 chunk:
// 8 threads per K-row, 128B (one cacheline) apart.
__device__ __forceinline__ void prefetch_b(const float* src, int ldsrc, int tid) {
    const float* p = src + (size_t)(tid >> 3) * ldsrc + (tid & 7) * 32;
    __builtin_prefetch(p, 0, 3);
}

// ---------------- zero y / comb / counts ----------------
__global__ __launch_bounds__(256) void moe_zero(float* __restrict__ y,
                                                float* __restrict__ comb,
                                                int* __restrict__ counts) {
    size_t i = (size_t)blockIdx.x * 256 + threadIdx.x;
    if (i < (size_t)T_TOK * DDIM) y[i] = 0.0f;
    if (i < (size_t)NEXP * T_TOK) comb[i] = 0.0f;
    if (i < NEXP) counts[i] = 0;
}

// ---------------- router: logits, softmax top-2, compaction ----------------
__global__ __launch_bounds__(256) void moe_router(const float* __restrict__ x,
                                                  const float* __restrict__ gate_w,
                                                  float* __restrict__ logits_out,
                                                  float* __restrict__ comb,
                                                  int* __restrict__ counts,
                                                  int* __restrict__ lists) {
    __shared__ float gw[DDIM * NEXP];
    for (int i = threadIdx.x; i < DDIM * NEXP; i += 256) gw[i] = gate_w[i];
    __syncthreads();

    const int t = blockIdx.x * 256 + threadIdx.x;
    const float* xr = x + (size_t)t * DDIM;

    float acc[NEXP];
#pragma unroll
    for (int e = 0; e < NEXP; ++e) acc[e] = 0.0f;

    for (int d = 0; d < DDIM; d += 4) {
        float4 xv = *(const float4*)(xr + d);
#pragma unroll
        for (int e = 0; e < NEXP; ++e) {
            acc[e] += xv.x * gw[(d + 0) * NEXP + e] + xv.y * gw[(d + 1) * NEXP + e] +
                      xv.z * gw[(d + 2) * NEXP + e] + xv.w * gw[(d + 3) * NEXP + e];
        }
    }

#pragma unroll
    for (int e = 0; e < NEXP; ++e) logits_out[(size_t)t * NEXP + e] = acc[e];

    // softmax (denominator cancels for top-2 renormalization)
    float m = acc[0];
#pragma unroll
    for (int e = 1; e < NEXP; ++e) m = fmaxf(m, acc[e]);
    float p[NEXP];
#pragma unroll
    for (int e = 0; e < NEXP; ++e) p[e] = __expf(acc[e] - m);

    int i1 = 0; float p1 = p[0];
#pragma unroll
    for (int e = 1; e < NEXP; ++e) if (p[e] > p1) { p1 = p[e]; i1 = e; }
    int i2 = (i1 == 0) ? 1 : 0; float p2 = p[i2];
#pragma unroll
    for (int e = 0; e < NEXP; ++e)
        if (e != i1 && p[e] > p2) { p2 = p[e]; i2 = e; }

    const float inv = __builtin_amdgcn_rcpf(p1 + p2);
    comb[(size_t)i1 * T_TOK + t] = p1 * inv;
    comb[(size_t)i2 * T_TOK + t] = p2 * inv;

    int pos1 = atomicAdd(&counts[i1], 1);
    lists[i1 * T_TOK + pos1] = t;
    int pos2 = atomicAdd(&counts[i2], 1);
    lists[i2 * T_TOK + pos2] = t;
}

// ---------------- slot offsets (8-entry exclusive prefix) ----------------
__global__ void moe_offsets(const int* __restrict__ counts, int* __restrict__ offs) {
    if (threadIdx.x == 0) {
        int s = 0;
#pragma unroll
        for (int e = 0; e < NEXP; ++e) { offs[e] = s; s += counts[e]; }
    }
}

// ---------------- FFN stage 1: h = silu(X@w1) * (X@w3) ----------------
__global__ __launch_bounds__(256) void moe_ffn1(const float* __restrict__ x,
                                                const float* __restrict__ w1,
                                                const float* __restrict__ w3,
                                                const int* __restrict__ counts,
                                                const int* __restrict__ offs,
                                                const int* __restrict__ lists,
                                                __bf16* __restrict__ hbuf) {
    const int e     = blockIdx.z;
    const int row0  = blockIdx.y * M_TILE;
    const int nbase = blockIdx.x * 256;
    const int cnt   = counts[e];
    if (row0 >= cnt) return;                     // uniform exit: EXEC stays all-ones
    const int rows  = min(M_TILE, cnt - row0);
    const int slot0 = offs[e] + row0;

    __shared__ __bf16 ldsX[M_TILE][DDIM];        // 64 KB, gathered token rows (bf16)
    __shared__ __bf16 ldsB1[256][32];            // 16 KB, w1 chunk, transposed [N][K]
    __shared__ __bf16 ldsB3[256][32];            // 16 KB, w3 chunk, transposed [N][K]

    const int tid = threadIdx.x;
    const float* w1e = w1 + (size_t)e * DDIM * FDIM + nbase;
    const float* w3e = w3 + (size_t)e * DDIM * FDIM + nbase;

    // prefetch first weight chunk while we stage X
    prefetch_b(w1e, FDIM, tid);
    prefetch_b(w3e, FDIM, tid);

    // preload 32xD token tile once: float4 global loads, packed 4xbf16 b64 LDS stores
#pragma unroll 4
    for (int j = 0; j < 32; ++j) {
        int idx = tid + 256 * j;                 // over 8192 float4s
        int r   = idx >> 8;                      // 256 float4 per row
        int c4  = idx & 255;
        float4 v = make_float4(0.f, 0.f, 0.f, 0.f);
        if (r < rows) {
            int tok = lists[e * T_TOK + row0 + r];
            v = *(const float4*)(x + (size_t)tok * DDIM + c4 * 4);
        }
        union { unsigned long long u; __bf16 bf[4]; } pk;
        pk.bf[0] = (__bf16)v.x; pk.bf[1] = (__bf16)v.y;
        pk.bf[2] = (__bf16)v.z; pk.bf[3] = (__bf16)v.w;
        *reinterpret_cast<unsigned long long*>(&ldsX[r][c4 * 4]) = pk.u;
    }
    __syncthreads();

    const int lane  = tid & 31;
    const int wv    = tid >> 5;
    const int lhalf = lane >> 4;                 // K-half select per ISA A/B layout
    const int lmod  = lane & 15;

    v8f acc1[2][2] = {{v8f{}, v8f{}}, {v8f{}, v8f{}}};
    v8f acc3[2][2] = {{v8f{}, v8f{}}, {v8f{}, v8f{}}};

    for (int kk = 0; kk < DDIM; kk += 32) {
        stage_b_t(w1e + (size_t)kk * FDIM, FDIM, ldsB1, tid);
        stage_b_t(w3e + (size_t)kk * FDIM, FDIM, ldsB3, tid);
        if (kk + 32 < DDIM) {                    // prefetch next K-chunk
            prefetch_b(w1e + (size_t)(kk + 32) * FDIM, FDIM, tid);
            prefetch_b(w3e + (size_t)(kk + 32) * FDIM, FDIM, tid);
        }
        __syncthreads();

        v16bf a[2];
#pragma unroll
        for (int mt = 0; mt < 2; ++mt)
            a[mt] = make_frag(&ldsX[mt * 16 + lmod][kk + 8 * lhalf],
                              &ldsX[mt * 16 + lmod][kk + 16 + 8 * lhalf]);
#pragma unroll
        for (int s = 0; s < 2; ++s) {
            const int n = wv * 32 + s * 16 + lmod;
            v16bf b1 = make_frag(&ldsB1[n][8 * lhalf], &ldsB1[n][16 + 8 * lhalf]);
#pragma unroll
            for (int mt = 0; mt < 2; ++mt)
                acc1[mt][s] = __builtin_amdgcn_wmma_f32_16x16x32_bf16(
                    false, a[mt], false, b1, (short)0, acc1[mt][s], false, false);
            v16bf b3 = make_frag(&ldsB3[n][8 * lhalf], &ldsB3[n][16 + 8 * lhalf]);
#pragma unroll
            for (int mt = 0; mt < 2; ++mt)
                acc3[mt][s] = __builtin_amdgcn_wmma_f32_16x16x32_bf16(
                    false, a[mt], false, b3, (short)0, acc3[mt][s], false, false);
        }
        __syncthreads();
    }

    // epilogue: h = silu(g) * u -> bf16; fast path when the whole tile is valid
    if (rows == M_TILE) {
#pragma unroll
        for (int mt = 0; mt < 2; ++mt)
#pragma unroll
            for (int s = 0; s < 2; ++s) {
                const int col = nbase + wv * 32 + s * 16 + lmod;
#pragma unroll
                for (int r = 0; r < 8; ++r) {
                    const int M = mt * 16 + r + 8 * lhalf;
                    float hv = fast_silu_mul(acc1[mt][s][r], acc3[mt][s][r]);
                    hbuf[(size_t)(slot0 + M) * FDIM + col] = (__bf16)hv;
                }
            }
    } else {
#pragma unroll
        for (int mt = 0; mt < 2; ++mt)
#pragma unroll
            for (int s = 0; s < 2; ++s) {
                const int col = nbase + wv * 32 + s * 16 + lmod;
#pragma unroll
                for (int r = 0; r < 8; ++r) {
                    const int M = mt * 16 + r + 8 * lhalf;
                    if (M < rows) {
                        float hv = fast_silu_mul(acc1[mt][s][r], acc3[mt][s][r]);
                        hbuf[(size_t)(slot0 + M) * FDIM + col] = (__bf16)hv;
                    }
                }
            }
    }
}

// ---------------- FFN stage 2: y += (h @ w2) * comb ----------------
__global__ __launch_bounds__(256) void moe_ffn2(const __bf16* __restrict__ hbuf,
                                                const float* __restrict__ w2,
                                                const int* __restrict__ counts,
                                                const int* __restrict__ offs,
                                                const int* __restrict__ lists,
                                                const float* __restrict__ comb,
                                                float* __restrict__ y) {
    const int e     = blockIdx.z;
    const int row0  = blockIdx.y * M_TILE;
    const int nbase = blockIdx.x * 256;          // over D
    const int cnt   = counts[e];
    if (row0 >= cnt) return;
    const int rows  = min(M_TILE, cnt - row0);
    const int slot0 = offs[e] + row0;

    __shared__ __bf16 ldsH[M_TILE][32];          // 2 KB per K-chunk
    __shared__ __bf16 ldsB[256][32];             // 16 KB, w2 chunk transposed [N][K]

    const int tid   = threadIdx.x;
    const int lane  = tid & 31;
    const int wv    = tid >> 5;
    const int lhalf = lane >> 4;
    const int lmod  = lane & 15;
    const float* w2e = w2 + (size_t)e * FDIM * DDIM + nbase;

    prefetch_b(w2e, DDIM, tid);

    v8f acc[2][2] = {{v8f{}, v8f{}}, {v8f{}, v8f{}}};

    for (int kk = 0; kk < FDIM; kk += 32) {
        // stage h tile (already bf16) 32x32: one 8-byte load per thread
        {
            int r  = tid >> 3;                   // 0..31
            int c4 = tid & 7;                    // 4 halves per thread
            unsigned long long v = 0ull;
            if (r < rows)
                v = *reinterpret_cast<const unsigned long long*>(
                        hbuf + (size_t)(slot0 + r) * FDIM + kk + c4 * 4);
            *reinterpret_cast<unsigned long long*>(&ldsH[r][c4 * 4]) = v;
        }
        stage_b_t(w2e + (size_t)kk * DDIM, DDIM, ldsB, tid);
        if (kk + 32 < FDIM)                      // prefetch next K-chunk
            prefetch_b(w2e + (size_t)(kk + 32) * DDIM, DDIM, tid);
        __syncthreads();

        v16bf a[2];
#pragma unroll
        for (int mt = 0; mt < 2; ++mt)
            a[mt] = make_frag(&ldsH[mt * 16 + lmod][8 * lhalf],
                              &ldsH[mt * 16 + lmod][16 + 8 * lhalf]);
#pragma unroll
        for (int s = 0; s < 2; ++s) {
            const int n = wv * 32 + s * 16 + lmod;
            v16bf b = make_frag(&ldsB[n][8 * lhalf], &ldsB[n][16 + 8 * lhalf]);
#pragma unroll
            for (int mt = 0; mt < 2; ++mt)
                acc[mt][s] = __builtin_amdgcn_wmma_f32_16x16x32_bf16(
                    false, a[mt], false, b, (short)0, acc[mt][s], false, false);
        }
        __syncthreads();
    }

    // epilogue: weighted scatter-add into y (exactly 2 contributions per element)
    if (rows == M_TILE) {
#pragma unroll
        for (int mt = 0; mt < 2; ++mt)
#pragma unroll
            for (int s = 0; s < 2; ++s) {
                const int col = nbase + wv * 32 + s * 16 + lmod;
#pragma unroll
                for (int r = 0; r < 8; ++r) {
                    const int M = mt * 16 + r + 8 * lhalf;
                    int   tok = lists[e * T_TOK + row0 + M];
                    float wgt = comb[(size_t)e * T_TOK + tok];
                    unsafeAtomicAdd(&y[(size_t)tok * DDIM + col], acc[mt][s][r] * wgt);
                }
            }
    } else {
#pragma unroll
        for (int mt = 0; mt < 2; ++mt)
#pragma unroll
            for (int s = 0; s < 2; ++s) {
                const int col = nbase + wv * 32 + s * 16 + lmod;
#pragma unroll
                for (int r = 0; r < 8; ++r) {
                    const int M = mt * 16 + r + 8 * lhalf;
                    if (M < rows) {
                        int   tok = lists[e * T_TOK + row0 + M];
                        float wgt = comb[(size_t)e * T_TOK + tok];
                        unsafeAtomicAdd(&y[(size_t)tok * DDIM + col], acc[mt][s][r] * wgt);
                    }
                }
            }
    }
}

extern "C" void kernel_launch(void* const* d_in, const int* in_sizes, int n_in,
                              void* d_out, int out_size, void* d_ws, size_t ws_size,
                              hipStream_t stream) {
    const float* x      = (const float*)d_in[0];
    const float* gate_w = (const float*)d_in[1];
    const float* w1     = (const float*)d_in[2];
    const float* w3     = (const float*)d_in[3];
    const float* w2     = (const float*)d_in[4];

    float* y_out      = (float*)d_out;                        // [T, D]
    float* logits_out = (float*)d_out + (size_t)T_TOK * DDIM; // [T, E]

    char* ws = (char*)d_ws;
    float*  comb   = (float*)(ws + COMB_OFF);
    int*    counts = (int*)(ws + CNT_OFF);
    int*    offs   = (int*)(ws + OFFS_OFF);
    int*    lists  = (int*)(ws + LIST_OFF);
    __bf16* hbuf   = (__bf16*)(ws + HBUF_OFF);

    moe_zero<<<(T_TOK * DDIM) / 256, 256, 0, stream>>>(y_out, comb, counts);
    moe_router<<<T_TOK / 256, 256, 0, stream>>>(x, gate_w, logits_out, comb, counts, lists);
    moe_offsets<<<1, 32, 0, stream>>>(counts, offs);
    moe_ffn1<<<dim3(FDIM / 256, T_TOK / M_TILE, NEXP), 256, 0, stream>>>(
        x, w1, w3, counts, offs, lists, hbuf);
    moe_ffn2<<<dim3(DDIM / 256, T_TOK / M_TILE, NEXP), 256, 0, stream>>>(
        hbuf, w2, counts, offs, lists, comb, y_out);
}
